// GraphAttentionLayer_55722905699137
// MI455X (gfx1250) — matrix operand order repaired and checked by reading, not compiled, not run on hardware
//
#include <hip/hip_runtime.h>
#include <hip/hip_bf16.h>
#include <stdint.h>

#define NN 50000
#define EE 400000
#define DD 128
#define DE 64
#define HH 8
#define SLOPE 0.2f
#define EPSLN 1e-5f

typedef __attribute__((ext_vector_type(16))) __bf16 v16bf;
typedef __attribute__((ext_vector_type(8)))  float  v8f;

// Packed B-fragment layout: frag[colTile][kStep][lane][j], j=0..15 contiguous.
// Element (k, c) with k = kStep*32 + (lane>>4)*16 + j, c = colTile*16 + (lane&15).

// ---------------- K0: zero accumulator + softmax sum scalar ----------------
__global__ void gat_zero_ws(float* __restrict__ accum, float* __restrict__ scalars, size_t n) {
    size_t i = (size_t)blockIdx.x * blockDim.x + threadIdx.x;
    size_t stride = (size_t)gridDim.x * blockDim.x;
    for (; i < n; i += stride) accum[i] = 0.0f;
    if (blockIdx.x == 0 && threadIdx.x == 0) scalars[1] = 0.0f;
}

// ---------------- K1: fold weights through A1, emit packed bf16 fragments --
// Logical Wc [128k x 512c]: c 0-127 = Wn, 128-255 = Wv, 256-383 = Wq@A1_q, 384-511 = Wk@A1_k
// Logical Wea [64k x 128c] = We@A1_e ; bfold[128] = b1 + bq@A1_q + bk@A1_k + be@A1_e
__global__ void gat_prep_weights(const float* __restrict__ Wn, const float* __restrict__ Wv,
                                 const float* __restrict__ Wq, const float* __restrict__ Wk,
                                 const float* __restrict__ We, const float* __restrict__ A1,
                                 const float* __restrict__ b1, const float* __restrict__ bq,
                                 const float* __restrict__ bk, const float* __restrict__ be,
                                 __bf16* __restrict__ WcP, __bf16* __restrict__ WeaP,
                                 float* __restrict__ bfold) {
    int t = blockIdx.x * blockDim.x + threadIdx.x;
    if (t < 32 * 4 * 32 * 16) {                     // packed Wc: 65536 elements
        int j     = t & 15;
        int lane  = (t >> 4) & 31;
        int kstep = (t >> 9) & 3;
        int ct    = t >> 11;
        int k = kstep * 32 + (lane >> 4) * 16 + j;  // 0..127
        int c = ct * 16 + (lane & 15);              // 0..511
        float v;
        if (c < 128)      v = Wn[k * 128 + c];
        else if (c < 256) v = Wv[k * 128 + (c - 128)];
        else if (c < 384) {
            int cc = c - 256; float s = 0.f;
            for (int q = 0; q < 128; ++q) s += Wq[k * 128 + q] * A1[q * 128 + cc];
            v = s;
        } else {
            int cc = c - 384; float s = 0.f;
            for (int q = 0; q < 128; ++q) s += Wk[k * 128 + q] * A1[(128 + q) * 128 + cc];
            v = s;
        }
        WcP[t] = (__bf16)v;
        return;
    }
    int t2 = t - 32 * 4 * 32 * 16;
    if (t2 >= 0 && t2 < 8 * 2 * 32 * 16) {          // packed Wea: 8192 elements
        int j     = t2 & 15;
        int lane  = (t2 >> 4) & 31;
        int kstep = (t2 >> 9) & 1;
        int ct    = t2 >> 10;
        int k = kstep * 32 + (lane >> 4) * 16 + j;  // 0..63
        int c = ct * 16 + (lane & 15);              // 0..127
        float s = 0.f;
        for (int q = 0; q < 128; ++q) s += We[k * 128 + q] * A1[(256 + q) * 128 + c];
        WeaP[t2] = (__bf16)s;
        return;
    }
    int t3 = t2 - 8 * 2 * 32 * 16;
    if (t3 >= 0 && t3 < 128) {
        float s = b1[t3];
        for (int q = 0; q < 128; ++q) {
            s += bq[q] * A1[q * 128 + t3]
               + bk[q] * A1[(128 + q) * 128 + t3]
               + be[q] * A1[(256 + q) * 128 + t3];
        }
        bfold[t3] = s;
    }
}

// ---------------- K2: fused node GEMM via WMMA -----------------------------
// [h | v | Qa | Ka] = X[N,128] @ Wc[128,512]  (bf16 in, f32 accum)
__global__ void gat_node_gemm(const float* __restrict__ X, const __bf16* __restrict__ WcP,
                              const float* __restrict__ bn, const float* __restrict__ bv,
                              float* __restrict__ Hout, float* __restrict__ Vout,
                              float* __restrict__ Qa, float* __restrict__ Ka) {
    __shared__ __bf16 sA[16 * 128];
    int rowBase = blockIdx.x * 16;
    int tid = threadIdx.x;
    // stage X tile as bf16 (float4 vectorized)
    {
        const float4* X4 = (const float4*)(X + (size_t)rowBase * 128);
        for (int i = tid; i < 16 * 32; i += 256) {
            float4 x = X4[i];
            int base = i * 4;
            sA[base + 0] = (__bf16)x.x;
            sA[base + 1] = (__bf16)x.y;
            sA[base + 2] = (__bf16)x.z;
            sA[base + 3] = (__bf16)x.w;
        }
    }
    __syncthreads();

    int wave = tid >> 5;
    int lane = tid & 31;
    int colTile = blockIdx.y * 8 + wave;           // 0..31
    int colBase = colTile * 16;
    int nlo = lane & 15;
    int hi  = lane >> 4;                           // 0 or 1

    const v16bf* Bp = (const v16bf*)WcP;
    v8f acc = {};
#pragma unroll
    for (int kstep = 0; kstep < 4; ++kstep) {
        int k0 = kstep * 32;
        v16bf a;
#pragma unroll
        for (int j = 0; j < 8; ++j) {
            a[j]     = sA[nlo * 128 + k0 + hi * 16 + j];
            a[j + 8] = sA[nlo * 128 + k0 + hi * 16 + 8 + j];
        }
        v16bf b = Bp[((colTile * 4 + kstep) << 5) + lane];   // one 32B load
        acc = __builtin_amdgcn_wmma_f32_16x16x32_bf16(false, a, false, b,
                                                      (short)0, acc, false, false);
    }

    int cg = colBase + nlo;
    float bias = 0.f;
    float* dst; int cl;
    if (cg < 128)      { dst = Hout; cl = cg;       bias = bn[cl]; }
    else if (cg < 256) { dst = Vout; cl = cg - 128; bias = bv[cl]; }
    else if (cg < 384) { dst = Qa;   cl = cg - 256; }
    else               { dst = Ka;   cl = cg - 384; }
#pragma unroll
    for (int r = 0; r < 8; ++r) {
        int m = r + hi * 8;
        dst[(size_t)(rowBase + m) * 128 + cl] = acc[r] + bias;
    }
}

// ---------------- K3: edge score kernel (fused WMMA + gather + heads) ------
__global__ void gat_edge_score(const float* __restrict__ EF, const __bf16* __restrict__ WeaP,
                               const float* __restrict__ Qa, const float* __restrict__ Ka,
                               const float* __restrict__ bfold, const float* __restrict__ A2,
                               const float* __restrict__ b2,
                               const int* __restrict__ srcIdx, const int* __restrict__ tgtIdx,
                               float* __restrict__ sOut) {
    __shared__ __bf16 sE[16 * 64];
    __shared__ float  sHM[16 * 128];
    __shared__ float  sA2[128 * 8];
    __shared__ float  sScore[16 * 8];
    __shared__ int    sSrc[16], sTgt[16];

    int eBase = blockIdx.x * 16;
    int tid = threadIdx.x;
    {
        const float4* E4 = (const float4*)(EF + (size_t)eBase * 64);
        for (int i = tid; i < 16 * 16; i += 256) {
            float4 x = E4[i];
            int base = i * 4;
            sE[base + 0] = (__bf16)x.x;
            sE[base + 1] = (__bf16)x.y;
            sE[base + 2] = (__bf16)x.z;
            sE[base + 3] = (__bf16)x.w;
        }
    }
    for (int i = tid; i < 128 * 8; i += 256) sA2[i] = A2[i];
    if (tid < 16) { sSrc[tid] = srcIdx[eBase + tid]; sTgt[tid] = tgtIdx[eBase + tid]; }
    __syncthreads();

    int wave = tid >> 5, lane = tid & 31;
    int nlo = lane & 15, hi = lane >> 4;
    int colBase = wave * 16;                        // wave == colTile (0..7)

    const v16bf* Bp = (const v16bf*)WeaP;
    v8f acc = {};
#pragma unroll
    for (int kstep = 0; kstep < 2; ++kstep) {
        int k0 = kstep * 32;
        v16bf a;
#pragma unroll
        for (int j = 0; j < 8; ++j) {
            a[j]     = sE[nlo * 64 + k0 + hi * 16 + j];
            a[j + 8] = sE[nlo * 64 + k0 + hi * 16 + 8 + j];
        }
        v16bf b = Bp[((wave * 2 + kstep) << 5) + lane];      // one 32B load
        acc = __builtin_amdgcn_wmma_f32_16x16x32_bf16(false, a, false, b,
                                                      (short)0, acc, false, false);
    }

    int d = colBase + nlo;
    float bf = bfold[d];
#pragma unroll
    for (int r = 0; r < 8; ++r) {
        int m = r + hi * 8;
        float v = acc[r] + bf + Qa[(size_t)sSrc[m] * 128 + d] + Ka[(size_t)sTgt[m] * 128 + d];
        v = v > 0.f ? v : v * SLOPE;
        sHM[m * 128 + d] = v;
    }
    __syncthreads();

    if (tid < 128) {                                // 16 edges x 8 heads
        int e = tid >> 3, h = tid & 7;
        float s = b2[h];
        for (int dd = 0; dd < 128; ++dd) s += sHM[e * 128 + dd] * sA2[dd * 8 + h];
        s = s > 0.f ? s : s * SLOPE;
        sScore[e * 8 + h] = s;
    }
    __syncthreads();
    if (tid < 16) {
        float s = 0.f;
        for (int h = 0; h < 8; ++h) s += sScore[tid * 8 + h];
        sOut[eBase + tid] = s * (1.0f / 8.0f);
    }
}

// ---------------- K4a/K4b/K4c: global softmax ------------------------------
__global__ void gat_reduce_max(const float* __restrict__ s, float* __restrict__ partial, int n) {
    __shared__ float sm[256];
    float m = -3.4e38f;
    for (int i = blockIdx.x * blockDim.x + threadIdx.x; i < n; i += gridDim.x * blockDim.x)
        m = fmaxf(m, s[i]);
    sm[threadIdx.x] = m;
    __syncthreads();
    for (int off = 128; off > 0; off >>= 1) {
        if ((int)threadIdx.x < off) sm[threadIdx.x] = fmaxf(sm[threadIdx.x], sm[threadIdx.x + off]);
        __syncthreads();
    }
    if (threadIdx.x == 0) partial[blockIdx.x] = sm[0];
}

__global__ void gat_final_max(const float* __restrict__ partial, int nb, float* __restrict__ scalars) {
    __shared__ float sm[256];
    float m = -3.4e38f;
    for (int i = threadIdx.x; i < nb; i += 256) m = fmaxf(m, partial[i]);
    sm[threadIdx.x] = m;
    __syncthreads();
    for (int off = 128; off > 0; off >>= 1) {
        if ((int)threadIdx.x < off) sm[threadIdx.x] = fmaxf(sm[threadIdx.x], sm[threadIdx.x + off]);
        __syncthreads();
    }
    if (threadIdx.x == 0) scalars[0] = sm[0];
}

__global__ void gat_sum_exp(const float* __restrict__ s, const float* __restrict__ scalars,
                            float* __restrict__ sumOut, int n) {
    __shared__ float sm[256];
    float mx = scalars[0];
    float acc = 0.f;
    for (int i = blockIdx.x * blockDim.x + threadIdx.x; i < n; i += gridDim.x * blockDim.x)
        acc += __expf(s[i] - mx);
    sm[threadIdx.x] = acc;
    __syncthreads();
    for (int off = 128; off > 0; off >>= 1) {
        if ((int)threadIdx.x < off) sm[threadIdx.x] += sm[threadIdx.x + off];
        __syncthreads();
    }
    if (threadIdx.x == 0) atomicAdd(sumOut, sm[0]);
}

// ---------------- K5: weighted scatter accum[src] += w * v[tgt] ------------
__global__ void gat_edge_scatter(const float* __restrict__ s, const float* __restrict__ scalars,
                                 const int* __restrict__ srcIdx, const int* __restrict__ tgtIdx,
                                 const float* __restrict__ V, float* __restrict__ accum) {
    int tid = blockIdx.x * blockDim.x + threadIdx.x;   // EE*32 threads
    int e = tid >> 5;
    if (e >= EE) return;
    int d = (tid & 31) * 4;
    float w = __expf(s[e] - scalars[0]) / scalars[1];
    const float4 v = *(const float4*)(V + (size_t)tgtIdx[e] * 128 + d);
    float* dst = accum + (size_t)srcIdx[e] * 128 + d;
    atomicAdd(dst + 0, w * v.x);
    atomicAdd(dst + 1, w * v.y);
    atomicAdd(dst + 2, w * v.z);
    atomicAdd(dst + 3, w * v.w);
}

// ---------------- K6: y = h + accum ; layernorm ----------------------------
__global__ void gat_finalize(const float* __restrict__ H, const float* __restrict__ accum,
                             const float* __restrict__ gamma, const float* __restrict__ beta,
                             float* __restrict__ out) {
    int node = blockIdx.x * 8 + (threadIdx.x >> 5);
    if (node >= NN) return;
    int lane = threadIdx.x & 31;
    const float4 h4 = ((const float4*)(H     + (size_t)node * 128))[lane];
    const float4 a4 = ((const float4*)(accum + (size_t)node * 128))[lane];
    float4 y;
    y.x = h4.x + a4.x; y.y = h4.y + a4.y; y.z = h4.z + a4.z; y.w = h4.w + a4.w;
    float sum   = y.x + y.y + y.z + y.w;
    float sumsq = y.x * y.x + y.y * y.y + y.z * y.z + y.w * y.w;
    for (int off = 16; off > 0; off >>= 1) {            // wave32 reduction
        sum   += __shfl_xor(sum,   off, 32);
        sumsq += __shfl_xor(sumsq, off, 32);
    }
    float mu  = sum * (1.0f / 128.0f);
    float var = sumsq * (1.0f / 128.0f) - mu * mu;
    float inv = rsqrtf(var + EPSLN);
    const float4 g = ((const float4*)gamma)[lane];
    const float4 b = ((const float4*)beta)[lane];
    float4 o;
    o.x = g.x * (y.x - mu) * inv + b.x;
    o.y = g.y * (y.y - mu) * inv + b.y;
    o.z = g.z * (y.z - mu) * inv + b.z;
    o.w = g.w * (y.w - mu) * inv + b.w;
    ((float4*)(out + (size_t)node * 128))[lane] = o;
}

// ---------------- launch ---------------------------------------------------
extern "C" void kernel_launch(void* const* d_in, const int* in_sizes, int n_in,
                              void* d_out, int out_size, void* d_ws, size_t ws_size,
                              hipStream_t stream) {
    const float* node_features = (const float*)d_in[0];
    const float* edge_features = (const float*)d_in[1];
    const float* Wn = (const float*)d_in[2];
    const float* bn = (const float*)d_in[3];
    const float* Wq = (const float*)d_in[4];
    const float* bq = (const float*)d_in[5];
    const float* Wk = (const float*)d_in[6];
    const float* bk = (const float*)d_in[7];
    const float* Wv = (const float*)d_in[8];
    const float* bv = (const float*)d_in[9];
    const float* We = (const float*)d_in[10];
    const float* be = (const float*)d_in[11];
    const float* A1 = (const float*)d_in[12];
    const float* b1 = (const float*)d_in[13];
    const float* A2 = (const float*)d_in[14];
    const float* b2 = (const float*)d_in[15];
    const float* gamma = (const float*)d_in[16];
    const float* beta  = (const float*)d_in[17];
    const int*   edge_index = (const int*)d_in[18];
    const int* srcIdx = edge_index;
    const int* tgtIdx = edge_index + EE;

    size_t nd = (size_t)NN * 128;
    float* Hout    = (float*)d_ws;
    float* Vout    = Hout + nd;
    float* Qa      = Vout + nd;
    float* Ka      = Qa + nd;
    float* accum   = Ka + nd;
    float* sVals   = accum + nd;
    float* partial = sVals + EE;
    float* scalars = partial + 512;
    // 64B-align the packed bf16 weight buffers (32B vector loads)
    uintptr_t p = (uintptr_t)(scalars + 8);
    p = (p + 63) & ~(uintptr_t)63;
    __bf16* WcP  = (__bf16*)p;
    __bf16* WeaP = WcP + 32 * 4 * 32 * 16;          // 65536 elements
    float*  bfold = (float*)(WeaP + 8 * 2 * 32 * 16); // after 8192 elements

    gat_zero_ws<<<4096, 256, 0, stream>>>(accum, scalars, nd);

    int prepThreads = 32 * 4 * 32 * 16 + 8 * 2 * 32 * 16 + 128;
    gat_prep_weights<<<(prepThreads + 255) / 256, 256, 0, stream>>>(
        Wn, Wv, Wq, Wk, We, A1, b1, bq, bk, be, WcP, WeaP, bfold);

    gat_node_gemm<<<dim3(NN / 16, 4), 256, 0, stream>>>(
        node_features, WcP, bn, bv, Hout, Vout, Qa, Ka);

    gat_edge_score<<<EE / 16, 256, 0, stream>>>(
        edge_features, WeaP, Qa, Ka, bfold, A2, b2, srcIdx, tgtIdx, sVals);

    gat_reduce_max<<<512, 256, 0, stream>>>(sVals, partial, EE);
    gat_final_max<<<1, 256, 0, stream>>>(partial, 512, scalars);
    gat_sum_exp<<<512, 256, 0, stream>>>(sVals, scalars, scalars + 1, EE);

    gat_edge_scatter<<<(EE * 32) / 256, 256, 0, stream>>>(
        sVals, scalars, srcIdx, tgtIdx, Vout, accum);

    gat_finalize<<<(NN + 7) / 8, 256, 0, stream>>>(
        Hout, accum, gamma, beta, (float*)d_out);
}
